// GlobalAttention_49039936586197
// MI455X (gfx1250) — matrix operand order repaired and checked by reading, not compiled, not run on hardware
//
#include <hip/hip_runtime.h>
#include <hip/hip_bf16.h>

// ---------------------------------------------------------------------------
// GlobalAttention on MI455X (gfx1250): f16 WMMA GEMMs + masked softmax.
//   q = query @ Wq^T ; e = q @ value^T ; scores = softmax(mask(e)) ; out = scores @ value
// Outputs (concat): outputs [8,2048,1024] f32, scores [8,2048,2048] f32.
// GEMM pipeline: global -> LDS via GLOBAL_LOAD_ASYNC_TO_LDS_B128 (ASYNCcnt,
// double buffered, 64-deep k-tiles) -> ds_load_b128 fragments
// -> v_wmma_f32_16x16x32_f16 (16 WMMAs per barrier pair).
// ---------------------------------------------------------------------------

typedef __attribute__((ext_vector_type(16))) _Float16 v16h;
typedef __attribute__((ext_vector_type(8)))  _Float16 v8h;
typedef __attribute__((ext_vector_type(4)))  _Float16 v4h;
typedef __attribute__((ext_vector_type(8)))  float    v8f;

#define BSZ 8
#define LQ  2048
#define LV  2048
#define QS  1024
#define VS  1024

#define TK      64   // k-depth of one staged LDS tile (2 WMMA k-steps)
#define LDS_LDA 72   // padded row stride (halves) for 64-half tile rows

// ---------------- conversion: f32 -> f16, 4 elements / thread --------------
__global__ __launch_bounds__(256) void cvt_f32_to_f16_v4(
    const float* __restrict__ s, _Float16* __restrict__ d, int n4) {
  int i = blockIdx.x * blockDim.x + threadIdx.x;
  if (i < n4) {
    float4 v = ((const float4*)s)[i];
    v4h h;
    h[0] = (_Float16)v.x; h[1] = (_Float16)v.y;
    h[2] = (_Float16)v.z; h[3] = (_Float16)v.w;
    ((v4h*)d)[i] = h;
  }
}

// -------- transpose + convert: value[b][m][d] f32 -> vt[b][d][m] f16 -------
__global__ __launch_bounds__(256) void transpose_cvt(
    const float* __restrict__ V, _Float16* __restrict__ VT) {
  __shared__ _Float16 tile[32][33];
  int b  = blockIdx.z;
  int m0 = blockIdx.y * 32;
  int d0 = blockIdx.x * 32;
  const float* Vb  = V  + (size_t)b * LV * VS;
  _Float16*    VTb = VT + (size_t)b * VS * LV;
  for (int i = threadIdx.y; i < 32; i += 8)
    tile[i][threadIdx.x] = (_Float16)Vb[(size_t)(m0 + i) * VS + d0 + threadIdx.x];
  __syncthreads();
  for (int i = threadIdx.y; i < 32; i += 8)
    VTb[(size_t)(d0 + i) * LV + m0 + threadIdx.x] = tile[threadIdx.x][i];
}

// ------------- async stage: 128x64 f16 tile, global -> LDS -----------------
// 16B chunks: 8 per row, 1024 total, 256 threads -> 4 async b128 per thread.
// LDS dest address for the async op = low 32 bits of the generic pointer
// (flat->LDS truncation, ISA 10.2).
__device__ __forceinline__ void stage_tile_async(
    const _Float16* __restrict__ g, int ld, int row0, int k0,
    _Float16* sm, int tid) {
#pragma unroll
  for (int it = 0; it < 4; ++it) {
    int id  = tid + it * 256;
    int row = id >> 3;
    int c   = id & 7;
    const void* gp = g + (size_t)(row0 + row) * ld + k0 + c * 8;
    unsigned    lp = (unsigned)(uintptr_t)(sm + row * LDS_LDA + c * 8);
    asm volatile("global_load_async_to_lds_b128 %0, %1, off"
                 :: "v"(lp), "v"(gp) : "memory");
  }
}

// ---------------- WMMA fragment loads from LDS (ISA 7.12.2) ----------------
// A: 16x32 f16. lane<16: row=lane, halves 0-7 -> K 0..7, 8-15 -> K 16..23;
//               lane>=16: row=lane-16, halves 0-7 -> K 8..15, 8-15 -> K 24..31.
__device__ __forceinline__ v16h lds_a_frag(const _Float16* sm, int m0, int kOff,
                                           int lane) {
  int r = lane & 15, hi = lane >> 4;
  const _Float16* p = sm + (m0 + r) * LDS_LDA + kOff + hi * 8;
  v8h c0 = *(const v8h*)p;
  v8h c1 = *(const v8h*)(p + 16);
  v16h f;
#pragma unroll
  for (int j = 0; j < 8; ++j) { f[j] = c0[j]; f[j + 8] = c1[j]; }
  return f;
}

// B: 32x16 f16 from B^T rows. lane<16: col=lane, halves 0-7 -> K 0..7, 8-15 -> K 16..23;
//                             lane>=16: col=lane-16, halves 0-7 -> K 8..15, 8-15 -> K 24..31.
__device__ __forceinline__ v16h lds_b_frag(const _Float16* sm, int n0, int kOff,
                                           int lane) {
  int n = lane & 15, hi = lane >> 4;
  const _Float16* p = sm + (n0 + n) * LDS_LDA + kOff + hi * 8;
  v8h c0 = *(const v8h*)p;
  v8h c1 = *(const v8h*)(p + 16);
  v16h f;
#pragma unroll
  for (int j = 0; j < 8; ++j) { f[j] = c0[j]; f[j + 8] = c1[j]; }
  return f;
}

// ------------------------------ batched GEMM -------------------------------
// C[M,N] = A[M,K] * Bt[N,K]^T, f16 in, f32 accumulate.
// Block: 256 threads = 8 waves, 2(M) x 4(N); wave tile 64x32 = 4x2 WMMA tiles.
// Block tile 128x128; 128x64 A/B k-tiles double buffered in LDS via async
// copies; 16 WMMAs per barrier pair. All dims divide the tiling exactly for
// this problem -> EXEC stays all-ones (WMMA requirement).
__global__ __launch_bounds__(256) void gemm_f16_nt(
    const _Float16* __restrict__ A, const _Float16* __restrict__ Bt,
    float* __restrict__ Cf, _Float16* __restrict__ Ch,
    int M, int N, int K,
    long long strideA, long long strideB, long long strideC) {
  __shared__ _Float16 smA[2][128 * LDS_LDA];
  __shared__ _Float16 smB[2][128 * LDS_LDA];

  int tid  = threadIdx.x;
  int lane = tid & 31;
  int wave = tid >> 5;
  int wm = wave & 1, wn = wave >> 1;
  int bm0 = blockIdx.y * 128;
  int bn0 = blockIdx.x * 128;

  const _Float16* Ab = A  + (size_t)blockIdx.z * strideA;
  const _Float16* Bb = Bt + (size_t)blockIdx.z * strideB;

  v8f acc[4][2] = {};

  // preload k-tile 0 into buffer 0
  stage_tile_async(Ab, K, bm0, 0, smA[0], tid);
  stage_tile_async(Bb, K, bn0, 0, smB[0], tid);

  int buf = 0;
  for (int k0 = 0; k0 < K; k0 += TK) {
    if (k0 + TK < K) {
      // kick off next k-tile into the other buffer, then wait for current
      stage_tile_async(Ab, K, bm0, k0 + TK, smA[buf ^ 1], tid);
      stage_tile_async(Bb, K, bn0, k0 + TK, smB[buf ^ 1], tid);
      asm volatile("s_wait_asynccnt 8" ::: "memory");
    } else {
      asm volatile("s_wait_asynccnt 0" ::: "memory");
    }
    __syncthreads();  // all waves' async writes for `buf` visible

#pragma unroll
    for (int kk = 0; kk < TK; kk += 32) {
      v16h a[4], b[2];
#pragma unroll
      for (int i = 0; i < 4; ++i)
        a[i] = lds_a_frag(smA[buf], wm * 64 + 16 * i, kk, lane);
#pragma unroll
      for (int j = 0; j < 2; ++j)
        b[j] = lds_b_frag(smB[buf], wn * 32 + 16 * j, kk, lane);
#pragma unroll
      for (int i = 0; i < 4; ++i)
#pragma unroll
        for (int j = 0; j < 2; ++j)
          acc[i][j] = __builtin_amdgcn_wmma_f32_16x16x32_f16(
              false, a[i], false, b[j], (short)0, acc[i][j], false, false);
    }

    __syncthreads();  // done reading `buf` before it is overwritten
    buf ^= 1;
  }

  // C/D layout: VGPR t -> M = t + (lane<16 ? 0 : 8); N = lane&15.
  int r = lane & 15, hi = lane >> 4;
  int bm = bm0 + wm * 64, bn = bn0 + wn * 32;
#pragma unroll
  for (int i = 0; i < 4; ++i)
#pragma unroll
    for (int j = 0; j < 2; ++j)
#pragma unroll
      for (int t = 0; t < 8; ++t) {
        int row = bm + 16 * i + hi * 8 + t;
        int col = bn + 16 * j + r;
        size_t idx = (size_t)blockIdx.z * strideC + (size_t)row * N + col;
        float v = acc[i][j][t];
        if (Cf) Cf[idx] = v;
        if (Ch) Ch[idx] = (_Float16)v;
      }
}

// ------------------------- masked row softmax ------------------------------
__global__ __launch_bounds__(256) void softmax_rows(
    float* __restrict__ S, _Float16* __restrict__ Sh,
    const int* __restrict__ mask) {
  __shared__ float red[256];
  int row = blockIdx.x;
  int b   = row >> 11;  // / LQ
  float*       srow = S  + (size_t)row * LV;
  _Float16*    hrow = Sh + (size_t)row * LV;
  const int*   mrow = mask + (size_t)b * LV;

  float x[8];
  float mx = -__builtin_inff();
#pragma unroll
  for (int t = 0; t < 8; ++t) {
    int c = threadIdx.x + t * 256;
    float v = srow[c];
    v = (mrow[c] != 0) ? v : -__builtin_inff();
    x[t] = v;
    mx = fmaxf(mx, v);
  }
  red[threadIdx.x] = mx;
  __syncthreads();
  for (int s = 128; s > 0; s >>= 1) {
    if (threadIdx.x < s)
      red[threadIdx.x] = fmaxf(red[threadIdx.x], red[threadIdx.x + s]);
    __syncthreads();
  }
  mx = red[0];
  __syncthreads();

  float sum = 0.f;
#pragma unroll
  for (int t = 0; t < 8; ++t) { float e = __expf(x[t] - mx); x[t] = e; sum += e; }
  red[threadIdx.x] = sum;
  __syncthreads();
  for (int s = 128; s > 0; s >>= 1) {
    if (threadIdx.x < s) red[threadIdx.x] += red[threadIdx.x + s];
    __syncthreads();
  }
  float inv = 1.0f / red[0];
#pragma unroll
  for (int t = 0; t < 8; ++t) {
    int c = threadIdx.x + t * 256;
    float p = x[t] * inv;
    srow[c] = p;
    hrow[c] = (_Float16)p;
  }
}

// ------------------------------- launcher ----------------------------------
extern "C" void kernel_launch(void* const* d_in, const int* in_sizes, int n_in,
                              void* d_out, int out_size, void* d_ws, size_t ws_size,
                              hipStream_t stream) {
  const float* query = (const float*)d_in[0];   // [8,2048,1024]
  const float* value = (const float*)d_in[1];   // [8,2048,1024]
  const int*   vmask = (const int*)d_in[2];     // [8,2048]
  const float* Wq    = (const float*)d_in[3];   // [1024,1024]

  float* outputs = (float*)d_out;                                // [8,2048,1024]
  float* scores  = outputs + (size_t)BSZ * LQ * VS;              // [8,2048,2048]

  char* ws = (char*)d_ws;
  _Float16* qh = (_Float16*)ws; ws += (size_t)BSZ * LQ * QS * 2; // query f16
  _Float16* wh = (_Float16*)ws; ws += (size_t)VS * QS * 2;       // Wq f16
  _Float16* vh = (_Float16*)ws; ws += (size_t)BSZ * LV * VS * 2; // value f16
  _Float16* vt = (_Float16*)ws; ws += (size_t)BSZ * VS * LV * 2; // value^T f16
  _Float16* ph = (_Float16*)ws; ws += (size_t)BSZ * LQ * VS * 2; // projected q f16
  _Float16* sh = (_Float16*)ws; ws += (size_t)BSZ * LQ * LV * 2; // scores f16

  // ---- conversions -------------------------------------------------------
  {
    int n4 = (BSZ * LQ * QS) / 4;
    cvt_f32_to_f16_v4<<<n4 / 256, 256, 0, stream>>>(query, qh, n4);
  }
  {
    int n4 = (VS * QS) / 4;
    cvt_f32_to_f16_v4<<<n4 / 256, 256, 0, stream>>>(Wq, wh, n4);
  }
  {
    int n4 = (BSZ * LV * VS) / 4;
    cvt_f32_to_f16_v4<<<n4 / 256, 256, 0, stream>>>(value, vh, n4);
  }
  transpose_cvt<<<dim3(VS / 32, LV / 32, BSZ), dim3(32, 8, 1), 0, stream>>>(value, vt);

  // ---- GEMM1: ph[16384,1024] = qh[16384,1024] @ wh[1024,1024]^T ----------
  gemm_f16_nt<<<dim3(VS / 128, (BSZ * LQ) / 128, 1), 256, 0, stream>>>(
      qh, wh, nullptr, ph, BSZ * LQ, VS, QS, 0, 0, 0);

  // ---- GEMM2 (per batch): e[2048,2048] = ph @ vh^T -> scores (f32) -------
  gemm_f16_nt<<<dim3(LV / 128, LQ / 128, BSZ), 256, 0, stream>>>(
      ph, vh, scores, nullptr, LQ, LV, VS,
      (long long)LQ * VS, (long long)LV * VS, (long long)LQ * LV);

  // ---- masked softmax over rows; write f32 scores + f16 copy -------------
  softmax_rows<<<BSZ * LQ, 256, 0, stream>>>(scores, sh, vmask);

  // ---- GEMM3 (per batch): out[2048,1024] = sh @ vt^T ---------------------
  gemm_f16_nt<<<dim3(VS / 128, LQ / 128, BSZ), 256, 0, stream>>>(
      sh, vt, outputs, nullptr, LQ, VS, LV,
      (long long)LQ * LV, (long long)VS * LV, (long long)LQ * VS);

  (void)in_sizes; (void)n_in; (void)out_size; (void)ws_size;
}
